// GraphAttentionWithTypedEdges_58626303591034
// MI455X (gfx1250) — matrix operand order repaired and checked by reading, not compiled, not run on hardware
//
#include <hip/hip_runtime.h>

// ---------------- Types & WMMA fragment helpers (gfx1250 wave32) ----------------

typedef __attribute__((ext_vector_type(16))) __bf16 v16bf;
typedef __attribute__((ext_vector_type(8)))  float  v8f;

union FragU { v16bf v; uint4 q[2]; };

// A-matrix fragment, 16x32 bf16 (MxK), row-major source with leading dim `ld`.
// ISA layout: lanes 0-15: row M=lane, K = k0+[0..7] and k0+[16..23]
//             lanes 16-31: row M=lane-16, K = k0+[8..15] and k0+[24..31]
static __device__ __forceinline__
v16bf load_frag_a(const __bf16* base, int ld, int m0, int k0, int lane) {
    const __bf16* p = base + (size_t)(m0 + (lane & 15)) * ld + k0 + ((lane >> 4) << 3);
    FragU f;
    f.q[0] = *(const uint4*)(p);
    f.q[1] = *(const uint4*)(p + 16);
    return f.v;
}

// B-matrix fragment, 32x16 bf16 (KxN). Source is "transposed" layout: row n holds
// the K-contiguous column n (leading dim `ld` over K).
// ISA layout: lane n (0-15): col N=n, K = k0+[0..7] and k0+[16..23]; lanes 16-31: K = k0+[8..15] and k0+[24..31]
static __device__ __forceinline__
v16bf load_frag_b(const __bf16* baseT, int ld, int n0, int k0, int lane) {
    const __bf16* p = baseT + (size_t)(n0 + (lane & 15)) * ld + k0 + ((lane >> 4) << 3);
    FragU f;
    f.q[0] = *(const uint4*)(p);
    f.q[1] = *(const uint4*)(p + 16);
    return f.v;
}

static __device__ __forceinline__
v8f wmma_bf16(v16bf a, v16bf b, v8f c) {
    return __builtin_amdgcn_wmma_f32_16x16x32_bf16(false, a, false, b, (short)0, c, false, false);
}

// Async global->LDS 16B copy (per active lane). ldsoff = LDS byte address
// (low 32 bits of the flat shared address), gaddr = 64-bit global address.
static __device__ __forceinline__
void async_copy_b128(unsigned int ldsoff, const void* gaddr) {
    asm volatile("global_load_async_to_lds_b128 %0, %1, off"
                 :: "v"(ldsoff), "v"((unsigned long long)(uintptr_t)gaddr)
                 : "memory");
}

// ---------------- Small prep kernels ----------------

// tb[t] = dot(edge_emb[t], type_w), t in [0,3)
__global__ void gawte_type_bias_kernel(const float* __restrict__ edge_emb,
                                       const float* __restrict__ type_w,
                                       float* __restrict__ tb) {
    __shared__ float red[256];
    int t = blockIdx.x;
    float s = 0.f;
    for (int i = threadIdx.x; i < 1024; i += 256)
        s += edge_emb[t * 1024 + i] * type_w[i];
    red[threadIdx.x] = s;
    __syncthreads();
    for (int off = 128; off > 0; off >>= 1) {
        if (threadIdx.x < off) red[threadIdx.x] += red[threadIdx.x + off];
        __syncthreads();
    }
    if (threadIdx.x == 0) tb[t] = red[0];
}

__global__ void gawte_zero_f32_kernel(float* __restrict__ p, int n) {
    for (int i = blockIdx.x * blockDim.x + threadIdx.x; i < n; i += gridDim.x * blockDim.x)
        p[i] = 0.f;
}

// bias_mat[src,dst] += tb[type]  (duplicates accumulate, matching scatter-add)
__global__ void gawte_scatter_kernel(const int* __restrict__ src, const int* __restrict__ dst,
                                     const int* __restrict__ types, const float* __restrict__ tb,
                                     float* __restrict__ bias_mat, int E) {
    int e = blockIdx.x * blockDim.x + threadIdx.x;
    if (e < E)
        atomicAdd(bias_mat + (size_t)src[e] * 2048 + dst[e], tb[types[e]]);
}

__global__ void gawte_cvt_bf16_kernel(const float* __restrict__ in, __bf16* __restrict__ out, int n) {
    for (int i = blockIdx.x * blockDim.x + threadIdx.x; i < n; i += gridDim.x * blockDim.x)
        out[i] = (__bf16)in[i];
}

// Wt[n][k] = (bf16) W[k][n], 1024x1024, tiled through LDS
__global__ void gawte_cvt_wT_kernel(const float* __restrict__ W, __bf16* __restrict__ Wt) {
    __shared__ float tile[16][17];
    int n0 = blockIdx.x * 16, k0 = blockIdx.y * 16;
    int tx = threadIdx.x & 15, ty = threadIdx.x >> 4;
    tile[ty][tx] = W[(size_t)(k0 + ty) * 1024 + n0 + tx];
    __syncthreads();
    Wt[(size_t)(n0 + ty) * 1024 + k0 + tx] = (__bf16)tile[tx][ty];
}

// ---------------- WMMA GEMM: C[M=4096,N=1024] = A[4096,1024] @ Wt^T + biasvec ----------------
// B slab (64 cols x 32 k) shared by all 8 waves is staged in LDS via async
// global->LDS copies, double-buffered on ASYNCcnt.
// MODE 0: f32 row-major out. MODE 1: bf16 row-major out. MODE 2: bf16 out transposed
//         per head for V: Vt[b][h][hd][T] with T=2048, H=16, hd=64.
template <int MODE>
__global__ __launch_bounds__(256)
void gawte_gemm_kernel(const __bf16* __restrict__ A, const __bf16* __restrict__ Bt,
                       const float* __restrict__ biasvec, void* __restrict__ Cout) {
    constexpr int K = 1024, N = 1024;
    __shared__ __align__(16) __bf16 bstage[2][64 * 32];  // 4KB per buffer

    const int tid = threadIdx.x;
    const int lane = tid & 31;
    const int wave = tid >> 5;
    const int m0 = blockIdx.x * 128 + wave * 16;   // 32 blocks in x
    const int n0 = blockIdx.y * 64;                // 16 blocks in y

    // This thread stages one 16B chunk of the 64x32 B slab: row=tid>>2, chunk=tid&3
    const int srow = tid >> 2, schunk = tid & 3;
    const __bf16* gsrc = Bt + (size_t)(n0 + srow) * K + schunk * 8;
    unsigned int ldso0 = (unsigned int)(uintptr_t)(&bstage[0][srow * 32 + schunk * 8]);
    unsigned int ldso1 = (unsigned int)(uintptr_t)(&bstage[1][srow * 32 + schunk * 8]);

    async_copy_b128(ldso0, gsrc);                  // prologue: k0 = 0 -> buffer 0

    v8f acc[4] = {};
    int it = 0;
    for (int k0 = 0; k0 < K; k0 += 32, ++it) {
        __syncthreads();  // all waves done reading the buffer we are about to overwrite
        if (k0 + 32 < K) {
            async_copy_b128((it & 1) ? ldso0 : ldso1, gsrc + k0 + 32);
            asm volatile("s_wait_asynccnt 1" ::: "memory");  // current buffer's copy done
        } else {
            asm volatile("s_wait_asynccnt 0" ::: "memory");
        }
        __syncthreads();  // every wave's staging is complete and visible

        const __bf16* lb = bstage[it & 1];
        v16bf a = load_frag_a(A, K, m0, k0, lane);
#pragma unroll
        for (int j = 0; j < 4; ++j) {
            v16bf b = load_frag_b(lb, 32, j * 16, 0, lane);
            acc[j] = wmma_bf16(a, b, acc[j]);
        }
    }

    const int row_in = (lane >> 4) << 3;   // 0 or 8
    const int col = lane & 15;
#pragma unroll
    for (int j = 0; j < 4; ++j) {
#pragma unroll
        for (int r = 0; r < 8; ++r) {
            int gm = m0 + row_in + r;
            int gn = n0 + j * 16 + col;
            float val = acc[j][r] + biasvec[gn];
            if (MODE == 0) {
                ((float*)Cout)[(size_t)gm * N + gn] = val;
            } else if (MODE == 1) {
                ((__bf16*)Cout)[(size_t)gm * N + gn] = (__bf16)val;
            } else {
                int bb = gm >> 11, t = gm & 2047;      // T = 2048
                int h = gn >> 6, hd = gn & 63;         // hd = 64
                ((__bf16*)Cout)[((size_t)((bb * 16 + h) * 64 + hd)) * 2048 + t] = (__bf16)val;
            }
        }
    }
}

// ---------------- Flash attention: one wave per (b, h, 16-row q tile) ----------------
// 64-key chunks: 16 score/PV WMMAs per chunk; row-sums via WMMA against an
// all-ones B fragment (lands broadcast in the C layout); only row-max uses
// 16-lane butterfly shuffles.
__global__ __launch_bounds__(256)
void gawte_attention_kernel(const __bf16* __restrict__ q, const __bf16* __restrict__ k,
                            const __bf16* __restrict__ vt, const float* __restrict__ bias_mat,
                            __bf16* __restrict__ attn_out) {
    constexpr int T = 2048, D = 1024, HD = 64;
    const float scale = 0.125f;  // 1/sqrt(64)

    __shared__ __align__(16) __bf16 lds_p[8 * 16 * 72];  // per-wave 16x64 P tile, stride 72
    const int lane = threadIdx.x & 31;
    const int wave = threadIdx.x >> 5;
    __bf16* lds_w = lds_p + wave * 16 * 72;

    const int unit = blockIdx.x * 8 + wave;  // 4096 units = 2*16*128
    const int qt = unit & 127;
    const int bh = unit >> 7;
    const int h = bh & 15;
    const int b = bh >> 4;
    const int q0 = qt * 16;

    const __bf16* qbase = q + (size_t)b * T * D + h * HD;            // row stride D
    const __bf16* kbase = k + (size_t)b * T * D + h * HD;            // row stride D
    const __bf16* vbase = vt + (size_t)(b * 16 + h) * HD * T;        // row (hd) stride T

    // q A-fragments for the two hd chunks (K = 0..31, 32..63)
    v16bf qa0 = load_frag_a(qbase, D, q0, 0, lane);
    v16bf qa1 = load_frag_a(qbase, D, q0, 32, lane);

    // all-ones bf16 B fragment for row-sum WMMA
    FragU onesf;
    onesf.q[0] = make_uint4(0x3F803F80u, 0x3F803F80u, 0x3F803F80u, 0x3F803F80u);
    onesf.q[1] = onesf.q[0];

    const int row_in = (lane >> 4) << 3;
    const int col = lane & 15;

    v8f acc[4] = {};
    v8f lacc = {};              // running row sums (broadcast across lanes via ones-WMMA)
    float m_run[8];
#pragma unroll
    for (int r = 0; r < 8; ++r) m_run[r] = -INFINITY;

    for (int kt = 0; kt < T; kt += 64) {
        // S = q @ k^T for 64 keys (four 16-wide tiles), K over hd = 64
        v8f s[4] = {};
#pragma unroll
        for (int c = 0; c < 4; ++c) {
            s[c] = wmma_bf16(qa0, load_frag_b(kbase, D, kt + c * 16, 0, lane), s[c]);
            s[c] = wmma_bf16(qa1, load_frag_b(kbase, D, kt + c * 16, 32, lane), s[c]);
        }

        // scale + typed-edge bias, online max, exp, stage P in LDS
#pragma unroll
        for (int r = 0; r < 8; ++r) {
            const float* brow = bias_mat + (size_t)(q0 + row_in + r) * T + kt + col;
            float v0 = s[0][r] * scale + brow[0];
            float v1 = s[1][r] * scale + brow[16];
            float v2 = s[2][r] * scale + brow[32];
            float v3 = s[3][r] * scale + brow[48];
            float mx = fmaxf(fmaxf(v0, v1), fmaxf(v2, v3));
            mx = fmaxf(mx, __shfl_xor(mx, 1, 32));
            mx = fmaxf(mx, __shfl_xor(mx, 2, 32));
            mx = fmaxf(mx, __shfl_xor(mx, 4, 32));
            mx = fmaxf(mx, __shfl_xor(mx, 8, 32));
            float mnew = fmaxf(m_run[r], mx);
            float alpha = __expf(m_run[r] - mnew);
            m_run[r] = mnew;
            float p0 = __expf(v0 - mnew);
            float p1 = __expf(v1 - mnew);
            float p2 = __expf(v2 - mnew);
            float p3 = __expf(v3 - mnew);
            lacc[r] *= alpha;
#pragma unroll
            for (int j = 0; j < 4; ++j) acc[j][r] *= alpha;
            __bf16* lrow = lds_w + (row_in + r) * 72 + col;
            lrow[0]  = (__bf16)p0;
            lrow[16] = (__bf16)p1;
            lrow[32] = (__bf16)p2;
            lrow[48] = (__bf16)p3;
        }

        // wave-local LDS ordering before re-reading P in A-fragment layout
        asm volatile("s_wait_dscnt 0" ::: "memory");
        v16bf pf0 = load_frag_a(lds_w, 72, 0, 0, lane);
        v16bf pf1 = load_frag_a(lds_w, 72, 0, 32, lane);

        // running row sums: lacc += P @ ones  (row sum broadcast across all lanes)
        lacc = wmma_bf16(pf0, onesf.v, lacc);
        lacc = wmma_bf16(pf1, onesf.v, lacc);

        // O += P @ V  (K = 64 keys, N = 64 hd)
#pragma unroll
        for (int j = 0; j < 4; ++j) {
            acc[j] = wmma_bf16(pf0, load_frag_b(vbase, T, j * 16, kt, lane), acc[j]);
            acc[j] = wmma_bf16(pf1, load_frag_b(vbase, T, j * 16, kt + 32, lane), acc[j]);
        }
    }

    // normalize and write bf16 output [B,T,D] at feature slice h*64
#pragma unroll
    for (int r = 0; r < 8; ++r) {
        float inv = 1.0f / lacc[r];
#pragma unroll
        for (int j = 0; j < 4; ++j) {
            attn_out[(size_t)(b * T + q0 + row_in + r) * D + h * HD + j * 16 + col] =
                (__bf16)(acc[j][r] * inv);
        }
    }
}

// ---------------- Host launch ----------------

extern "C" void kernel_launch(void* const* d_in, const int* in_sizes, int n_in,
                              void* d_out, int out_size, void* d_ws, size_t ws_size,
                              hipStream_t stream) {
    const float* x        = (const float*)d_in[0];
    const float* Wq       = (const float*)d_in[1];
    const float* bq       = (const float*)d_in[2];
    const float* Wk       = (const float*)d_in[3];
    const float* bk       = (const float*)d_in[4];
    const float* Wv       = (const float*)d_in[5];
    const float* bv       = (const float*)d_in[6];
    const float* Wo       = (const float*)d_in[7];
    const float* bo       = (const float*)d_in[8];
    const float* edge_emb = (const float*)d_in[9];
    const float* type_w   = (const float*)d_in[10];
    const int*   e_src    = (const int*)d_in[11];
    const int*   e_dst    = (const int*)d_in[12];
    const int*   e_typ    = (const int*)d_in[13];
    const int E = in_sizes[11];

    const size_t T = 2048, D = 1024, MB = 4096;  // MB = B*T rows

    size_t off = 0;
    auto alloc = [&](size_t bytes) {
        void* p = (char*)d_ws + off;
        off += (bytes + 255) & ~(size_t)255;
        return p;
    };
    float*  bias_mat = (float*)alloc(T * T * sizeof(float));      // 16 MB
    float*  tb       = (float*)alloc(3 * sizeof(float));
    __bf16* xb       = (__bf16*)alloc(MB * D * 2);                // 8 MB
    __bf16* wqT      = (__bf16*)alloc(D * D * 2);
    __bf16* wkT      = (__bf16*)alloc(D * D * 2);
    __bf16* wvT      = (__bf16*)alloc(D * D * 2);
    __bf16* woT      = (__bf16*)alloc(D * D * 2);
    __bf16* qb       = (__bf16*)alloc(MB * D * 2);
    __bf16* kb       = (__bf16*)alloc(MB * D * 2);
    __bf16* vtb      = (__bf16*)alloc(MB * D * 2);                // [b][h][64][2048]
    __bf16* aob      = (__bf16*)alloc(MB * D * 2);
    (void)ws_size; (void)n_in; (void)out_size;

    // 1) per-type scalar bias, dense bias matrix
    gawte_type_bias_kernel<<<3, 256, 0, stream>>>(edge_emb, type_w, tb);
    gawte_zero_f32_kernel<<<2048, 256, 0, stream>>>(bias_mat, (int)(T * T));
    gawte_scatter_kernel<<<(E + 255) / 256, 256, 0, stream>>>(e_src, e_dst, e_typ, tb, bias_mat, E);

    // 2) bf16 conversions (weights transposed to [N][K])
    gawte_cvt_bf16_kernel<<<2048, 256, 0, stream>>>(x, xb, (int)(MB * D));
    dim3 tgrid(64, 64);
    gawte_cvt_wT_kernel<<<tgrid, 256, 0, stream>>>(Wq, wqT);
    gawte_cvt_wT_kernel<<<tgrid, 256, 0, stream>>>(Wk, wkT);
    gawte_cvt_wT_kernel<<<tgrid, 256, 0, stream>>>(Wv, wvT);
    gawte_cvt_wT_kernel<<<tgrid, 256, 0, stream>>>(Wo, woT);

    // 3) QKV projections (WMMA); V stored transposed per head
    dim3 ggrid(32, 16);
    gawte_gemm_kernel<1><<<ggrid, 256, 0, stream>>>(xb, wqT, bq, qb);
    gawte_gemm_kernel<1><<<ggrid, 256, 0, stream>>>(xb, wkT, bk, kb);
    gawte_gemm_kernel<2><<<ggrid, 256, 0, stream>>>(xb, wvT, bv, vtb);

    // 4) flash attention with typed-edge bias
    gawte_attention_kernel<<<512, 256, 0, stream>>>(qb, kb, vtb, bias_mat, aob);

    // 5) output projection -> f32 d_out
    gawte_gemm_kernel<0><<<ggrid, 256, 0, stream>>>(aob, woT, bo, (float*)d_out);
}
